// PointAggregator_25056839205897
// MI455X (gfx1250) — matrix operand order repaired and checked by reading, not compiled, not run on hardware
//
#include <hip/hip_runtime.h>
#include <hip/hip_bf16.h>

// ---------------------------------------------------------------------------
// Types for WMMA fragments (gfx1250, wave32)
// ---------------------------------------------------------------------------
typedef _Float16 half8  __attribute__((ext_vector_type(8)));
typedef _Float16 half16 __attribute__((ext_vector_type(16)));
typedef float    floatx8 __attribute__((ext_vector_type(8)));

union Frag {
    half16 v;
    half8  h[2];
};

#define NSAMP   163840      // B*R*SR = 1*4096*40
#define FDIM    32
#define HDIM    256
#define MT      64          // rows per block in MLP kernel
#define STRIDE  264         // padded LDS row stride in halves (16B aligned)
#define EPS     1e-8f

// ---------------------------------------------------------------------------
// Pack a row-major f32 weight [kSteps*32, 256] into f16 WMMA B-fragment order.
// Fragment (kk, n): lane holds 16 contiguous halves = B[k0+i][n*16 + lane%16],
// k0 = kk*32 + (lane>=16 ? 16 : 0).  Linear index:
//   t = ((kk*16 + n)*32 + lane)*16 + i
// ---------------------------------------------------------------------------
__global__ void pack_w_kernel(const float* __restrict__ W,
                              _Float16* __restrict__ dst, int kSteps)
{
    int t = blockIdx.x * 256 + threadIdx.x;
    int total = kSteps * 16 * 32 * 16;
    if (t >= total) return;
    int i    = t & 15;
    int lane = (t >> 4) & 31;
    int n    = (t >> 9) & 15;
    int kk   = t >> 13;
    int row  = kk * 32 + ((lane & 16) ? 16 : 0) + i;
    int col  = n * 16 + (lane & 15);
    dst[t] = (_Float16)W[row * HDIM + col];
}

// ---------------------------------------------------------------------------
// Aggregation: one wave (32 lanes) per sample.
//   w_k = mask_k / max(|d_k|^2, eps);  w /= max(sum w, eps)
//   feat[f] = sum_k w_k * emb[k][f]    (f = lane)
// ---------------------------------------------------------------------------
__global__ __launch_bounds__(256)
void feat_kernel(const float* __restrict__ emb,
                 const float* __restrict__ dists,
                 const int*   __restrict__ mask,
                 _Float16*    __restrict__ feat)
{
    const int wave = threadIdx.x >> 5;
    const int lane = threadIdx.x & 31;
    const long s = (long)blockIdx.x * 8 + wave;
    if (s >= NSAMP) return;

    float w = 0.0f;
    if (lane < 8) {
        const float* dp = dists + s * 24 + lane * 3;
        float dx = dp[0], dy = dp[1], dz = dp[2];
        float d2 = dx * dx + dy * dy + dz * dz;
        float m = (float)mask[s * 8 + lane];
        w = m / fmaxf(d2, EPS);
    }
    float wk[8];
    float wsum = 0.0f;
#pragma unroll
    for (int k = 0; k < 8; ++k) {
        wk[k] = __shfl(w, k, 32);
        wsum += wk[k];
    }
    const float inv = 1.0f / fmaxf(wsum, EPS);

    const float* ep = emb + s * 256 + lane;   // emb[s][k][lane], k stride 32
    float f = 0.0f;
#pragma unroll
    for (int k = 0; k < 8; ++k)
        f += wk[k] * ep[k * 32];

    feat[s * 32 + lane] = (_Float16)(f * inv);
}

// ---------------------------------------------------------------------------
// One hidden 256->256 layer for a single wave's 16-row slice.
// A fragments loaded once (8 x v16h in regs), then 16 N-tiles x 8 K-steps
// of v_wmma_f32_16x16x32_f16, bias (LDS-staged) in accumulator init,
// ReLU on writeback.  Each wave reads/writes only its own rows =>
// no block barrier needed between layers.
// ---------------------------------------------------------------------------
__device__ __forceinline__ void hidden_layer(
    _Float16* act, const _Float16* __restrict__ pW,
    const float* bias /* LDS */,
    int rowBase, int lane, int col, int hi)
{
    Frag a[8];
    const _Float16* arow = act + (rowBase + col) * STRIDE;
#pragma unroll
    for (int kk = 0; kk < 8; ++kk) {
        a[kk].h[0] = *(const half8*)(arow + kk * 32 + hi * 16);
        a[kk].h[1] = *(const half8*)(arow + kk * 32 + hi * 16 + 8);
    }
#pragma unroll 4
    for (int n = 0; n < 16; ++n) {
        const float bv = bias[n * 16 + col];       // LDS broadcast-ish load
        floatx8 c;
#pragma unroll
        for (int j = 0; j < 8; ++j) c[j] = bv;
#pragma unroll
        for (int kk = 0; kk < 8; ++kk) {
            Frag b;
            const _Float16* bp = pW + (((kk * 16 + n) * 32 + lane) << 4);
            b.h[0] = *(const half8*)(bp);
            b.h[1] = *(const half8*)(bp + 8);
            c = __builtin_amdgcn_wmma_f32_16x16x32_f16(
                    false, a[kk].v, false, b.v, (short)0, c, false, false);
        }
#pragma unroll
        for (int j = 0; j < 8; ++j) {
            const float rv = c[j] > 0.0f ? (float)c[j] : 0.0f;
            act[(rowBase + j + hi * 8) * STRIDE + n * 16 + col] = (_Float16)rv;
        }
    }
}

// ---------------------------------------------------------------------------
// Fused MLP: 2560 blocks x 128 threads (4 waves), 64 rows per block.
// ---------------------------------------------------------------------------
__global__ __launch_bounds__(128)
void mlp_kernel(const _Float16* __restrict__ feat,
                const _Float16* __restrict__ pW1a,
                const _Float16* __restrict__ pW1b,
                const _Float16* __restrict__ pW3a,
                const _Float16* __restrict__ pW3b,
                const float* __restrict__ b1a, const float* __restrict__ b1b,
                const float* __restrict__ b3a, const float* __restrict__ b3b,
                const float* __restrict__ Wa,  const float* __restrict__ ba,
                const float* __restrict__ Wc,  const float* __restrict__ bc,
                const float* __restrict__ viewdirs,
                float* __restrict__ out)
{
    __shared__ _Float16 act[MT * STRIDE];   // 33792 B
    __shared__ float sBias[4 * HDIM];       // 4096 B : b1a|b1b|b3a|b3b
    __shared__ float sWa[HDIM];             // 1024 B
    __shared__ float sWc[280 * 3];          // 3360 B

    const int tid  = threadIdx.x;
    const int wave = tid >> 5;
    const int lane = tid & 31;
    const int col  = lane & 15;
    const int hi   = lane >> 4;
    const int rowBase = wave * 16;
    const long gBase  = (long)blockIdx.x * MT;

    // stage biases + head weights into LDS (keeps LOADcnt clean in GEMM loops)
    for (int i = tid; i < HDIM; i += 128) {
        sBias[i]            = b1a[i];
        sBias[HDIM + i]     = b1b[i];
        sBias[2 * HDIM + i] = b3a[i];
        sBias[3 * HDIM + i] = b3b[i];
        sWa[i] = Wa[i];
    }
    for (int i = tid; i < 840; i += 128) sWc[i] = Wc[i];
    __syncthreads();

    // ---- Layer 1: feat(K=32) -> 256, ReLU. A frag straight from global feat.
    {
        const _Float16* frow = feat + (gBase + rowBase + col) * 32;
        Frag a;
        a.h[0] = *(const half8*)(frow + hi * 16);
        a.h[1] = *(const half8*)(frow + hi * 16 + 8);
#pragma unroll
        for (int n = 0; n < 16; ++n) {
            const float bv = sBias[n * 16 + col];
            floatx8 c;
#pragma unroll
            for (int j = 0; j < 8; ++j) c[j] = bv;
            Frag b;
            const _Float16* bp = pW1a + ((n * 32 + lane) << 4);
            b.h[0] = *(const half8*)(bp);
            b.h[1] = *(const half8*)(bp + 8);
            c = __builtin_amdgcn_wmma_f32_16x16x32_f16(
                    false, a.v, false, b.v, (short)0, c, false, false);
#pragma unroll
            for (int j = 0; j < 8; ++j) {
                const float rv = c[j] > 0.0f ? (float)c[j] : 0.0f;
                act[(rowBase + j + hi * 8) * STRIDE + n * 16 + col] = (_Float16)rv;
            }
        }
    }

    // ---- Layers 2..4 (256->256 each, ReLU)
    hidden_layer(act, pW1b, sBias + HDIM,     rowBase, lane, col, hi);
    hidden_layer(act, pW3a, sBias + 2 * HDIM, rowBase, lane, col, hi);
    hidden_layer(act, pW3b, sBias + 3 * HDIM, rowBase, lane, col, hi);

    __syncthreads();

    // ---- Heads: alpha = softplus(h.Wa + ba - 1); color = sigmoid(..)*1.002-1e-3
    // All 128 threads: 2 threads per row, each handles 128 of the 256 dims
    // with vectorized half8 LDS loads; partials combined via __shfl_xor.
    {
        const int r    = tid >> 1;    // row 0..63
        const int part = tid & 1;     // which half of the hidden dim
        const _Float16* hrow = act + r * STRIDE + part * 128;
        const float* wa = sWa + part * 128;
        const float* wc = sWc + part * 128 * 3;

        float aacc = 0.0f, c0 = 0.0f, c1 = 0.0f, c2 = 0.0f;
#pragma unroll
        for (int ch = 0; ch < 16; ++ch) {
            const half8 hv8 = *(const half8*)(hrow + ch * 8);
#pragma unroll
            for (int j = 0; j < 8; ++j) {
                const int i = ch * 8 + j;
                const float hv = (float)hv8[j];
                aacc += hv * wa[i];
                c0 += hv * wc[i * 3 + 0];
                c1 += hv * wc[i * 3 + 1];
                c2 += hv * wc[i * 3 + 2];
            }
        }
        aacc += __shfl_xor(aacc, 1, 32);
        c0   += __shfl_xor(c0, 1, 32);
        c1   += __shfl_xor(c1, 1, 32);
        c2   += __shfl_xor(c2, 1, 32);

        if (part == 0) {
            const long g = gBase + r;
            aacc += ba[0];
            c0 += bc[0]; c1 += bc[1]; c2 += bc[2];

            // view-direction positional encoding (4 freqs, sin then cos per freq)
            float vx = viewdirs[g * 3 + 0];
            float vy = viewdirs[g * 3 + 1];
            float vz = viewdirs[g * 3 + 2];
            const float nrm = sqrtf(vx * vx + vy * vy + vz * vz);
            const float inv = 1.0f / fmaxf(nrm, EPS);
            float v3[3] = {vx * inv, vy * inv, vz * inv};
#pragma unroll
            for (int f = 0; f < 4; ++f) {
                const float fr = (float)(1 << f);
#pragma unroll
                for (int k = 0; k < 3; ++k) {
                    const float ang = v3[k] * fr;
                    const float sv = __sinf(ang);
                    const float cv = __cosf(ang);
                    const int js = 256 + f * 6 + k;      // sin slot
                    const int jc = js + 3;               // cos slot
                    c0 += sv * sWc[js * 3 + 0] + cv * sWc[jc * 3 + 0];
                    c1 += sv * sWc[js * 3 + 1] + cv * sWc[jc * 3 + 1];
                    c2 += sv * sWc[js * 3 + 2] + cv * sWc[jc * 3 + 2];
                }
            }

            const float x = aacc - 1.0f;
            const float alpha = (x > 20.0f) ? x : log1pf(__expf(x));
            const float s0 = 1.0f / (1.0f + __expf(-c0)) * 1.002f - 1e-3f;
            const float s1 = 1.0f / (1.0f + __expf(-c1)) * 1.002f - 1e-3f;
            const float s2 = 1.0f / (1.0f + __expf(-c2)) * 1.002f - 1e-3f;

            float4 o = make_float4(alpha, s0, s1, s2);
            *(float4*)(out + g * 4) = o;
        }
    }
}

// ---------------------------------------------------------------------------
// Host launch
// ---------------------------------------------------------------------------
extern "C" void kernel_launch(void* const* d_in, const int* in_sizes, int n_in,
                              void* d_out, int out_size, void* d_ws, size_t ws_size,
                              hipStream_t stream)
{
    const float* emb      = (const float*)d_in[0];   // [1,4096,40,8,32]
    const float* dists    = (const float*)d_in[1];   // [1,4096,40,8,3]
    const float* viewdirs = (const float*)d_in[2];   // [1,4096,40,3]
    const int*   mask     = (const int*)  d_in[3];   // [1,4096,40,8]
    const float* W1a = (const float*)d_in[4];
    const float* b1a = (const float*)d_in[5];
    const float* W1b = (const float*)d_in[6];
    const float* b1b = (const float*)d_in[7];
    const float* W3a = (const float*)d_in[8];
    const float* b3a = (const float*)d_in[9];
    const float* W3b = (const float*)d_in[10];
    const float* b3b = (const float*)d_in[11];
    const float* Wa  = (const float*)d_in[12];
    const float* ba  = (const float*)d_in[13];
    const float* Wc  = (const float*)d_in[14];
    const float* bc  = (const float*)d_in[15];
    float* out = (float*)d_out;

    // workspace carve-up (all 256B aligned)
    char* ws = (char*)d_ws;
    _Float16* feat = (_Float16*)ws;                      ws += (size_t)NSAMP * 32 * 2;
    _Float16* pW1a = (_Float16*)ws;                      ws += 8192 * 2;
    _Float16* pW1b = (_Float16*)ws;                      ws += 65536 * 2;
    _Float16* pW3a = (_Float16*)ws;                      ws += 65536 * 2;
    _Float16* pW3b = (_Float16*)ws;                      ws += 65536 * 2;

    // 1) pack weights into WMMA B-fragment layout (f16)
    pack_w_kernel<<<32,  256, 0, stream>>>(W1a, pW1a, 1);
    pack_w_kernel<<<256, 256, 0, stream>>>(W1b, pW1b, 8);
    pack_w_kernel<<<256, 256, 0, stream>>>(W3a, pW3a, 8);
    pack_w_kernel<<<256, 256, 0, stream>>>(W3b, pW3b, 8);

    // 2) neighbor aggregation -> f16 features  (163840 samples, 8 per block)
    feat_kernel<<<NSAMP / 8, 256, 0, stream>>>(emb, dists, mask, feat);

    // 3) fused WMMA MLP + heads  (64 rows per block)
    mlp_kernel<<<NSAMP / MT, 128, 0, stream>>>(
        feat, pW1a, pW1b, pW3a, pW3b,
        b1a, b1b, b3a, b3b, Wa, ba, Wc, bc, viewdirs, out);
}
